// Gpt2Attention_41936060678518
// MI455X (gfx1250) — matrix operand order repaired and checked by reading, not compiled, not run on hardware
//
#include <hip/hip_runtime.h>
#include <hip/hip_bf16.h>

// ---------------------------------------------------------------------------
// Types for CDNA5 WMMA (wave32): D(16x16 f32) = A(16x32 bf16) x B(32x16 bf16) + C
// ---------------------------------------------------------------------------
typedef __attribute__((ext_vector_type(16))) __bf16 v16bf;
typedef __attribute__((ext_vector_type(8)))  float  v8f;
typedef __attribute__((ext_vector_type(4)))  unsigned int u32x4;
typedef __attribute__((ext_vector_type(4)))  int          i32x4;
typedef __attribute__((ext_vector_type(8)))  int          i32x8;

union BF16x16 { v16bf v; unsigned int u[8]; };

__device__ __forceinline__ unsigned short f2bf(float f) {
    union { float f; unsigned int u; } c; c.f = f;
    unsigned int u = c.u;
    u += 0x7fffu + ((u >> 16) & 1u);      // round-to-nearest-even
    return (unsigned short)(u >> 16);
}

// LDS byte offset of a generic pointer aliasing LDS (flat->LDS keeps offset
// in the low 32 bits).
__device__ __forceinline__ unsigned lds_off(const void* p) {
    return (unsigned)(unsigned long long)p;
}

// Per-lane async DMA: 16 bytes global -> LDS, tracked by ASYNCcnt.
__device__ __forceinline__ void async_copy_b128(const void* lds_dst, const void* gsrc) {
    unsigned           l = lds_off(lds_dst);
    unsigned long long g = (unsigned long long)gsrc;
    asm volatile("global_load_async_to_lds_b128 %0, %1, off"
                 :: "v"(l), "v"(g) : "memory");
}

// A fragment (16x32 bf16, MxK). src is a [16 x >=32] row-major bf16 region.
// K order used here (contiguous, same as B loaders): lanes 0-15 (M=lane): K=0..15; lanes 16-31: K=16..31.
__device__ __forceinline__ v16bf load_a_frag(const unsigned short* src, int stride) {
    const int lane = threadIdx.x & 31;
    const int m    = lane & 15;
    const int koff = (lane >> 4) << 4;     // 0 or 16
    const unsigned int* p = (const unsigned int*)(src + (size_t)m * stride);
    BF16x16 r;
#pragma unroll
    for (int g = 0; g < 4; ++g) {
        r.u[g]     = p[(koff >> 1) + g];
        r.u[4 + g] = p[(koff >> 1) + 4 + g];
    }
    return r.v;
}

// B fragment from K-major storage: "row" n holds its 32 K-values contiguously.
// K order used here (contiguous, same as A loader): lanes 0-15 (N=lane): K=0..15; lanes 16-31: K=16..31.
__device__ __forceinline__ v16bf load_b_frag(const unsigned short* src, int stride) {
    const int lane = threadIdx.x & 31;
    const int n    = lane & 15;
    const int koff = (lane >> 4) << 4;     // 0 or 16
    const unsigned int* p = (const unsigned int*)(src + (size_t)n * stride + koff);
    BF16x16 r;
#pragma unroll
    for (int g = 0; g < 8; ++g) r.u[g] = p[g];
    return r.v;
}

// B fragment from natural row-major [k][n] storage (strided ds_load_u16 gather).
// Used only off the hot path (flash-attention P@V, 8 fragments per key block).
__device__ __forceinline__ v16bf load_b_frag_rowmajor(const unsigned short* lds,
                                                      int ldn, int ncol) {
    const int lane = threadIdx.x & 31;
    const int n    = ncol + (lane & 15);
    const int koff = (lane >> 4) << 4;     // 0 or 16
    BF16x16 r;
#pragma unroll
    for (int g = 0; g < 8; ++g) {
        unsigned lo = lds[(size_t)(koff + 2 * g)     * ldn + n];
        unsigned hi = lds[(size_t)(koff + 2 * g + 1) * ldn + n];
        r.u[g] = lo | (hi << 16);
    }
    return r.v;
}

__device__ __forceinline__ float half_reduce_max(float v) {
#pragma unroll
    for (int m = 8; m >= 1; m >>= 1) v = fmaxf(v, __shfl_xor(v, m, 32));
    return v;
}
__device__ __forceinline__ float half_reduce_sum(float v) {
#pragma unroll
    for (int m = 8; m >= 1; m >>= 1) v += __shfl_xor(v, m, 32);
    return v;
}

// ---------------------------------------------------------------------------
// fp32 -> bf16 bulk conversion
// ---------------------------------------------------------------------------
__global__ void __launch_bounds__(256)
f32_to_bf16(const float* __restrict__ in, unsigned short* __restrict__ out, long long n)
{
    long long i = (long long)blockIdx.x * blockDim.x + threadIdx.x;
    long long stride = (long long)gridDim.x * blockDim.x;
    for (; i < n; i += stride) out[i] = f2bf(in[i]);
}

// ---------------------------------------------------------------------------
// GEMM: C[M,N](fp32) = A[M,K](bf16) * B[K,N](bf16). Tiles 128x128x32, 8 waves.
// A tile: TDM tensor_load_to_lds (fallback: manual copy). B tile: batched
// coalesced global loads + transpose-on-store into [n][k] LDS so the inner
// loop reads clean ds_load_b128 fragments. Double-buffered, 1 barrier/K-step.
// ---------------------------------------------------------------------------
#define GBM 128
#define GBN 128
#define GBK 32

__device__ __forceinline__ void stage_tiles(unsigned short* Asb, unsigned short* Bsb,
                                            const unsigned short* Ag,
                                            const unsigned short* Bg,
                                            int M, int N, int K,
                                            int m0, int n0, int k0, int tid)
{
    // ---- B tile: batch-issue 8 coalesced dword loads, then 16 b16 stores ----
    const unsigned short* brow = Bg + (size_t)k0 * N + n0;
    unsigned vbuf[8];
#pragma unroll
    for (int it = 0; it < 8; ++it) {          // 2048 dwords / 256 threads
        int c  = tid + it * 256;
        int k  = c >> 6;                      // 64 dwords per k-row
        int n2 = (c & 63) * 2;
        vbuf[it] = *(const unsigned*)(brow + (size_t)k * N + n2);
    }
#pragma unroll
    for (int it = 0; it < 8; ++it) {
        int c  = tid + it * 256;
        int k  = c >> 6;
        int n2 = (c & 63) * 2;
        Bsb[(size_t)(n2)     * GBK + k] = (unsigned short)(vbuf[it] & 0xffffu);
        Bsb[(size_t)(n2 + 1) * GBK + k] = (unsigned short)(vbuf[it] >> 16);
    }

    // ---- A tile: whole 128x32 bf16 tile via the Tensor Data Mover ----
#if __has_builtin(__builtin_amdgcn_tensor_load_to_lds)
    if (tid < 32) {      // one wave issues the DMA
        unsigned long long ga  = (unsigned long long)(const void*)
                                 (Ag + (size_t)m0 * K + k0);
        unsigned long long st0 = (unsigned long long)K;       // dim0 stride (elems)
        u32x4 g0;
        g0[0] = 1u;                                           // count=1, user mode
        g0[1] = lds_off(Asb);                                 // lds_addr
        g0[2] = (unsigned)(ga & 0xffffffffu);                 // global_addr lo
        g0[3] = (unsigned)((ga >> 32) & 0x01ffffffu) | (2u << 30);  // hi | type=2
        i32x8 g1;
        g1[0] = 0x10000;                                      // data_size=1 (2B)
        g1[1] = (int)(((unsigned)K & 0xffffu) << 16);         // tensor_dim0 lo16
        g1[2] = (int)((((unsigned)K >> 16) & 0xffffu) |
                      (((unsigned)M & 0xffffu) << 16));       // td0 hi | td1 lo
        g1[3] = (int)((((unsigned)M >> 16) & 0xffffu) |
                      (32u << 16));                           // td1 hi | tile_dim0=32
        g1[4] = 128;                                          // tile_dim1=128, td2=0
        g1[5] = (int)(st0 & 0xffffffffu);                     // dim0 stride lo
        g1[6] = (int)((st0 >> 32) & 0xffffu);                 // dim0 stride hi
        g1[7] = 0;
        i32x4 z4 = {};
#if __clang_major__ >= 23
        i32x8 z8 = {};
        __builtin_amdgcn_tensor_load_to_lds(g0, g1, z4, z4, z8, 0);
#else
        __builtin_amdgcn_tensor_load_to_lds(g0, g1, z4, z4, 0);
#endif
    }
#else
    // fallback: manual copy, 2048 dwords
#pragma unroll
    for (int it = 0; it < 8; ++it) {
        int c = tid + it * 256;
        int m = c >> 4, p = c & 15;
        ((unsigned*)Asb)[m * 16 + p] =
            *(const unsigned*)(Ag + (size_t)(m0 + m) * K + k0 + p * 2);
    }
#endif

    // speculative prefetch of the tile after next (global_prefetch_b8)
    if (k0 + 2 * GBK < K)
        __builtin_prefetch(Bg + (size_t)(k0 + 2 * GBK) * N + n0 + (tid & 15) * 8, 0, 1);
}

__global__ void __launch_bounds__(256)
gemm_bf16_wmma(const unsigned short* __restrict__ A, const unsigned short* __restrict__ Bm,
               float* __restrict__ C, int M, int N, int K)
{
    __shared__ unsigned short As[2][GBM * GBK];   // [m][k]
    __shared__ unsigned short Bs[2][GBN * GBK];   // [n][k]  (transposed)

    const int tid = threadIdx.x;
    const int wv  = tid >> 5;
    const int wr  = wv >> 1;              // 0..3 : 32-row band
    const int wc  = wv & 1;               // 0..1 : 64-col band
    const int m0  = blockIdx.y * GBM;
    const int n0  = blockIdx.x * GBN;

    v8f vz = {};
    v8f acc[2][4];
#pragma unroll
    for (int i = 0; i < 2; ++i)
#pragma unroll
        for (int j = 0; j < 4; ++j) acc[i][j] = vz;

    stage_tiles(As[0], Bs[0], A, Bm, M, N, K, m0, n0, 0, tid);

    int buf = 0;
    for (int k0 = 0; k0 < K; k0 += GBK, buf ^= 1) {
#if __has_builtin(__builtin_amdgcn_tensor_load_to_lds)
        if (tid < 32) __builtin_amdgcn_s_wait_tensorcnt(0);   // my A tile landed
#endif
        __syncthreads();                   // everyone's A DMA + B stores visible
        if (k0 + GBK < K)                  // stage next tile into other buffer
            stage_tiles(As[buf ^ 1], Bs[buf ^ 1], A, Bm, M, N, K,
                        m0, n0, k0 + GBK, tid);

        v16bf af[2], bfr[4];
#pragma unroll
        for (int i = 0; i < 2; ++i)
            af[i] = load_a_frag(As[buf] + (size_t)(wr * 32 + i * 16) * GBK, GBK);
#pragma unroll
        for (int j = 0; j < 4; ++j)
            bfr[j] = load_b_frag(Bs[buf] + (size_t)(wc * 64 + j * 16) * GBK, GBK);
#pragma unroll
        for (int i = 0; i < 2; ++i)
#pragma unroll
            for (int j = 0; j < 4; ++j)
                acc[i][j] = __builtin_amdgcn_wmma_f32_16x16x32_bf16(
                    false, af[i], false, bfr[j], (short)0, acc[i][j], false, false);
    }

    const int lane = tid & 31;
    const int col  = lane & 15;
    const int rb   = (lane >> 4) * 8;
#pragma unroll
    for (int i = 0; i < 2; ++i)
#pragma unroll
        for (int j = 0; j < 4; ++j)
#pragma unroll
            for (int e = 0; e < 8; ++e) {
                int m = m0 + wr * 32 + i * 16 + rb + e;
                int n = n0 + wc * 64 + j * 16 + col;
                C[(size_t)m * N + n] = acc[i][j][e];
            }
}

// ---------------------------------------------------------------------------
// fused RMS-norm (eps=0.01) + NeoX RoPE + bf16 pack.
// qkv fp32 [B,S,3,H,D] -> Qb/Kb/Vb bf16 [B,H,S,D]. One wave per (b,s,h).
// ---------------------------------------------------------------------------
__global__ void __launch_bounds__(128)
norm_rope_pack(const float* __restrict__ qkv,
               const float* __restrict__ qw, const float* __restrict__ kw,
               unsigned short* __restrict__ Qb, unsigned short* __restrict__ Kb,
               unsigned short* __restrict__ Vb, int B, int S, int H, int D)
{
    const int row  = blockIdx.x * 4 + (threadIdx.x >> 5);
    const int lane = threadIdx.x & 31;
    if (row >= B * S * H) return;
    const int h = row % H;
    const int s = (row / H) % S;
    const int b = row / (H * S);

    const size_t base = (((size_t)(b * S + s) * 3) * H + h) * (size_t)D;
    const size_t khd  = (size_t)H * D;

    float xq[4], xk[4], xv[4];
#pragma unroll
    for (int j = 0; j < 4; ++j) {
        int d = lane + j * 32;
        xq[j] = qkv[base + 0 * khd + d];
        xk[j] = qkv[base + 1 * khd + d];
        xv[j] = qkv[base + 2 * khd + d];
    }
    float sq = 0.f, sk = 0.f;
#pragma unroll
    for (int j = 0; j < 4; ++j) { sq += xq[j] * xq[j]; sk += xk[j] * xk[j]; }
#pragma unroll
    for (int m = 16; m >= 1; m >>= 1) {
        sq += __shfl_xor(sq, m, 32);
        sk += __shfl_xor(sk, m, 32);
    }
    const float rq = rsqrtf(sq * (1.f / 128.f) + 0.01f);
    const float rk = rsqrtf(sk * (1.f / 128.f) + 0.01f);

    const size_t obase = (((size_t)b * H + h) * S + s) * (size_t)D;
#pragma unroll
    for (int j = 0; j < 4; ++j) {
        int d  = lane + j * 32;
        int pj = j ^ 2;                                // partner element: d ^ 64
        float nq  = xq[j]  * rq * qw[d];
        float nk  = xk[j]  * rk * kw[d];
        float nqp = xq[pj] * rq * qw[d ^ 64];
        float nkp = xk[pj] * rk * kw[d ^ 64];
        float rotq = (d < 64) ? -nqp : nqp;
        float rotk = (d < 64) ? -nkp : nkp;
        int   i    = d & 63;
        float freq = __expf(-(float)i * (9.210340371976184f / 64.f)); // 10000^(-i/64)
        float ang  = (float)s * freq;
        float cs = __cosf(ang), sn = __sinf(ang);
        Qb[obase + d] = f2bf(nq * cs + rotq * sn);
        Kb[obase + d] = f2bf(nk * cs + rotk * sn);
        Vb[obase + d] = f2bf(xv[j]);
    }
}

// ---------------------------------------------------------------------------
// causal flash attention, bf16 WMMA, fp32 online softmax. Writes bf16 output.
// Grid: (S/64, B*H); 4 waves/WG; 1 wave = 16 query rows; key blocks of 32.
// V tile is async-DMA'd raw [key][d] into LDS (coalesced, no VGPR round-trip).
// ---------------------------------------------------------------------------
#define NEG_BIG (-1e30f)

__global__ void __launch_bounds__(128)
flash_attn_kernel(const unsigned short* __restrict__ Qb,
                  const unsigned short* __restrict__ Kb,
                  const unsigned short* __restrict__ Vb,
                  unsigned short* __restrict__ Ob, int B, int S, int H, int D)
{
    const int lane = threadIdx.x & 31;
    const int w    = threadIdx.x >> 5;
    const int bh   = blockIdx.y;
    const int b    = bh / H, h = bh % H;
    const int q0   = blockIdx.x * 64 + w * 16;
    const float scale = 0.08838834764831845f;     // 1/sqrt(128)

    __shared__ unsigned short Vr[32 * 128];        // raw V tile [key][d]
    __shared__ unsigned short Ps[4][16 * 32];      // per-wave P scratch [row][key]

    const unsigned short* Qrow = Qb + ((size_t)bh * S + q0) * D;
    v16bf aq[4];
#pragma unroll
    for (int c = 0; c < 4; ++c) aq[c] = load_a_frag(Qrow + c * 32, D);

    v8f vz = {};
    v8f o_acc[8];
#pragma unroll
    for (int j = 0; j < 8; ++j) o_acc[j] = vz;
    float m_e[8], l_e[8];
#pragma unroll
    for (int e = 0; e < 8; ++e) { m_e[e] = NEG_BIG; l_e[e] = 0.f; }

    const int col   = lane & 15;
    const int rbase = (lane >> 4) * 8;
    const int nkb   = (blockIdx.x * 64 + 63) / 32 + 1;   // uniform across the WG

    for (int kb = 0; kb < nkb; ++kb) {
        __syncthreads();   // previous tile fully consumed by all waves
        {   // async DMA the raw 8KB V tile: 512 x 16B chunks, 4 per thread
            const unsigned short* Vsrc = Vb + ((size_t)bh * S + kb * 32) * D;
#pragma unroll
            for (int it = 0; it < 4; ++it) {
                int c = threadIdx.x + it * 128;
                async_copy_b128(Vr + (size_t)c * 8, Vsrc + (size_t)c * 8);
            }
        }
        if (kb + 1 < nkb) {   // prefetch next key/value block
            size_t nb = ((size_t)bh * S + (kb + 1) * 32) * D + (threadIdx.x & 127) * 8;
            __builtin_prefetch(Kb + nb, 0, 1);
            __builtin_prefetch(Vb + nb, 0, 1);
        }
        asm volatile("s_wait_asynccnt 0" ::: "memory");  // my chunks landed
        __syncthreads();                                 // everyone's landed

        // scores: two 16x16 tiles (keys kb*32..+15, +16..+31), K direct from L2
        v8f s0 = vz, s1 = vz;
        const unsigned short* Krow = Kb + ((size_t)bh * S + kb * 32) * D;
#pragma unroll
        for (int c = 0; c < 4; ++c) {
            v16bf b0 = load_b_frag(Krow + c * 32, D);
            v16bf b1 = load_b_frag(Krow + (size_t)16 * D + c * 32, D);
            s0 = __builtin_amdgcn_wmma_f32_16x16x32_bf16(false, aq[c], false, b0,
                                                         (short)0, s0, false, false);
            s1 = __builtin_amdgcn_wmma_f32_16x16x32_bf16(false, aq[c], false, b1,
                                                         (short)0, s1, false, false);
        }

        // causal mask + online softmax per accumulator row
        float alpha[8];
#pragma unroll
        for (int e = 0; e < 8; ++e) {
            int   qg  = q0 + rbase + e;
            int   k0g = kb * 32 + col;
            float v0  = s0[e] * scale, v1 = s1[e] * scale;
            if (k0g > qg)       v0 = NEG_BIG;
            if (k0g + 16 > qg)  v1 = NEG_BIG;
            float mnew = fmaxf(m_e[e], half_reduce_max(fmaxf(v0, v1)));
            float a    = __expf(m_e[e] - mnew);
            float p0   = __expf(v0 - mnew), p1 = __expf(v1 - mnew);
            l_e[e]   = l_e[e] * a + half_reduce_sum(p0 + p1);
            m_e[e]   = mnew;
            alpha[e] = a;
            Ps[w][(rbase + e) * 32 + col]      = f2bf(p0);
            Ps[w][(rbase + e) * 32 + col + 16] = f2bf(p1);
        }
        asm volatile("s_wait_dscnt 0" ::: "memory");   // wave-local P store->load

        // O = diag(alpha)*O + P @ V   (V fragments gathered from raw [key][d])
        v16bf ap = load_a_frag(&Ps[w][0], 32);
#pragma unroll
        for (int j = 0; j < 8; ++j) {
            v8f t = o_acc[j];
#pragma unroll
            for (int e = 0; e < 8; ++e) t[e] *= alpha[e];
            v16bf bv = load_b_frag_rowmajor(Vr, 128, j * 16);
            o_acc[j] = __builtin_amdgcn_wmma_f32_16x16x32_bf16(false, ap, false, bv,
                                                               (short)0, t, false, false);
        }
    }

    // normalize and write bf16 O in [B,S,H,D] (A-operand of output projection)
#pragma unroll
    for (int j = 0; j < 8; ++j)
#pragma unroll
        for (int e = 0; e < 8; ++e) {
            int q = q0 + rbase + e;
            Ob[(((size_t)b * S + q) * H + h) * D + j * 16 + col] =
                f2bf(o_acc[j][e] / l_e[e]);
        }
}

// ---------------------------------------------------------------------------
// Host-side orchestration
// ---------------------------------------------------------------------------
extern "C" void kernel_launch(void* const* d_in, const int* in_sizes, int n_in,
                              void* d_out, int out_size, void* d_ws, size_t ws_size,
                              hipStream_t stream)
{
    (void)in_sizes; (void)n_in; (void)out_size; (void)ws_size;
    const int B = 4, S = 2048, E = 2048, H = 16, D = 128;

    const float* x      = (const float*)d_in[0];
    const float* w_qkv  = (const float*)d_in[1];   // [E, 3, H, D] == [E, 6144]
    const float* w_proj = (const float*)d_in[2];   // [H, D, E]   == [2048, 2048]
    const float* qnw    = (const float*)d_in[3];
    const float* knw    = (const float*)d_in[4];
    float*       out    = (float*)d_out;           // [B, S, E]

    const size_t MS = (size_t)B * S;               // 8192 tokens
    char* ws = (char*)d_ws;
    size_t off = 0;
    float*          qkv = (float*)(ws + off);          off += MS * 3 * H * D * sizeof(float);
    unsigned short* xbf = (unsigned short*)(ws + off); off += MS * (size_t)E * 2;
    unsigned short* wqb = (unsigned short*)(ws + off); off += (size_t)E * 3 * H * D * 2;
    unsigned short* wpb = (unsigned short*)(ws + off); off += (size_t)H * D * E * 2;
    unsigned short* Qb  = (unsigned short*)(ws + off); off += MS * (size_t)H * D * 2;
    unsigned short* Kb  = (unsigned short*)(ws + off); off += MS * (size_t)H * D * 2;
    unsigned short* Vb  = (unsigned short*)(ws + off); off += MS * (size_t)H * D * 2;
    unsigned short* Obf = (unsigned short*)(ws + off); off += MS * (size_t)H * D * 2;

    // 0) one-time bf16 conversion of all GEMM operands (halves HBM traffic)
    f32_to_bf16<<<2048, 256, 0, stream>>>(x,      xbf, (long long)MS * E);
    f32_to_bf16<<<2048, 256, 0, stream>>>(w_qkv,  wqb, (long long)E * 3 * H * D);
    f32_to_bf16<<<2048, 256, 0, stream>>>(w_proj, wpb, (long long)H * D * E);

    // 1) QKV projection: [8192 x 2048] @ [2048 x 6144] -> qkv fp32 [B,S,3,H,D]
    gemm_bf16_wmma<<<dim3(3 * H * D / GBN, (int)MS / GBM), 256, 0, stream>>>(
        xbf, wqb, qkv, (int)MS, 3 * H * D, E);

    // 2) RMS-norm + RoPE + bf16 pack into [B,H,S,D]
    norm_rope_pack<<<dim3(B * S * H / 4), 128, 0, stream>>>(
        qkv, qnw, knw, Qb, Kb, Vb, B, S, H, D);

    // 3) causal flash attention -> Obf bf16 [B,S,H,D]
    flash_attn_kernel<<<dim3(S / 64, B * H), 128, 0, stream>>>(
        Qb, Kb, Vb, Obf, B, S, H, D);

    // 4) output projection: [8192 x 2048] @ [2048 x 2048] -> out fp32
    gemm_bf16_wmma<<<dim3(E / GBN, (int)MS / GBM), 256, 0, stream>>>(
        Obf, wpb, out, (int)MS, E, H * D);
}
